// CVRPModel_84911503442120
// MI455X (gfx1250) — compile-verified, matrix operand order checked by reading, not run on hardware
//
#include <hip/hip_runtime.h>

// MI455X / gfx1250 implementation. All GEMMs use v_wmma_f32_16x16x32_bf16
// (bf16 A/B, f32 accumulate): at 8x the MAC density of f32 WMMA this pushes
// the whole pipeline into the memory-bound regime (~73 FLOP/B needed vs
// ~215 FLOP/B machine balance at 23.3 TB/s), while f32 accumulation +
// f32 transcendental math preserve the reference numerics.
//
// Round-2 changes: native __bf16 casts (hardware v_cvt, replacing a 3-VALU
// software RNE pack per element) + global_prefetch_b8 on the HBM streams.

#define DEV static __device__ __forceinline__

typedef __attribute__((ext_vector_type(16))) __bf16 v16bf;
typedef __attribute__((ext_vector_type(8)))  float  v8f;

constexpr int B_ = 16, P_ = 200, N_ = 2000, D_ = 128;
constexpr int NPAD = 2048;           // padded N for transposed bf16 buffers
constexpr int PTILES = 13;           // ceil(200/16)
constexpr float INV_SQRT_D = 1.0f / 11.313708498984761f;
constexpr float LOGIT_CLIP = 10.0f;

DEV __bf16 f2bf(float x) { return (__bf16)x; }   // hw cvt (v_cvt*_bf16_f32)
DEV unsigned short f2bfu(float x) {
  union { __bf16 b; unsigned short s; } o;
  o.b = (__bf16)x;
  return o.s;
}

DEV v8f wmma_bf16(v16bf a, v16bf b, v8f c) {
  // 8 args: (neg_a, A, neg_b, B, c_mod, C, reuse_a, reuse_b)
  return __builtin_amdgcn_wmma_f32_16x16x32_bf16(
      false, a, false, b, (short)0, c, false, false);
}

// Fragment element i holds K = (i&7) + 8*kg + (i>=8 ? 16 : 0)  (ISA 16-bit
// 16x32 layout; kg = lane>=16). Source: 32 consecutive f32 along K.
DEV v16bf frag_f32(const float* rowbase, int kg) {
  const float* g0 = rowbase + 8 * kg;
  const float* g1 = rowbase + 16 + 8 * kg;
  v16bf a;
#pragma unroll
  for (int i = 0; i < 8; ++i) a[i] = f2bf(g0[i]);
#pragma unroll
  for (int i = 0; i < 8; ++i) a[8 + i] = f2bf(g1[i]);
  return a;
}

// Same K pattern from 32 consecutive bf16 (two 16B vector loads per lane).
DEV v16bf frag_bf16(const unsigned short* rowbase, int kg) {
  union { v16bf v; uint4 q[2]; } u;
  u.q[0] = *(const uint4*)(rowbase + 8 * kg);
  u.q[1] = *(const uint4*)(rowbase + 16 + 8 * kg);
  return u.v;
}

DEV void load8(const float* p, float* o) {
  union { float4 v[2]; float f[8]; } u;
  u.v[0] = *(const float4*)p;
  u.v[1] = *(const float4*)(p + 4);
#pragma unroll
  for (int i = 0; i < 8; ++i) o[i] = u.f[i];
}

// ---------------------------------------------------------------------------
// Kernel 1: k = nodes @ Wk^T, v = nodes @ Wv^T; store exp(k), exp(k)*v
// transposed [b][d][n] in bf16 (B-operand-ready), plus nodes in bf16 row-major.
// grid (125, B), block 128 (4 waves); wave w owns d-tiles {2w, 2w+1}.
// ---------------------------------------------------------------------------
__global__ void __launch_bounds__(128) prep_kv_kernel(
    const float* __restrict__ nodes, const float* __restrict__ Wk,
    const float* __restrict__ Wv, unsigned short* __restrict__ expkT,
    unsigned short* __restrict__ ekvT, unsigned short* __restrict__ nodesbf) {
  const int nt = blockIdx.x, b = blockIdx.y;
  const int n0 = nt * 16;
  const int tid = threadIdx.x;
  const int lane = tid & 31, w = tid >> 5;
  const int col = lane & 15, hi = lane >> 4;

  // bf16 row-major copy of this node tile
  for (int i = tid; i < 16 * D_; i += 128) {
    int m = i >> 7, d = i & 127;
    size_t idx = ((size_t)(b * N_ + n0 + m)) * D_ + d;
    nodesbf[idx] = f2bfu(nodes[idx]);
  }

  const float* arow = nodes + ((size_t)(b * N_ + n0 + col)) * D_;
  const int dA = 2 * w * 16 + col, dB = dA + 16;
  const float* wkA = Wk + (size_t)dA * D_;
  const float* wkB = Wk + (size_t)dB * D_;
  const float* wvA = Wv + (size_t)dA * D_;
  const float* wvB = Wv + (size_t)dB * D_;

  v8f kC0 = {}, kC1 = {}, vC0 = {}, vC1 = {};
#pragma unroll
  for (int kc = 0; kc < 4; ++kc) {
    const int k0 = kc * 32;
    v16bf a = frag_f32(arow + k0, hi);
    kC0 = wmma_bf16(a, frag_f32(wkA + k0, hi), kC0);
    kC1 = wmma_bf16(a, frag_f32(wkB + k0, hi), kC1);
    vC0 = wmma_bf16(a, frag_f32(wvA + k0, hi), vC0);
    vC1 = wmma_bf16(a, frag_f32(wvB + k0, hi), vC1);
  }

  // C layout: vgpr r, lane -> (M = r + 8*hi, N = col). Store transposed:
  // expkT[(b*128+d)*NPAD + n], rows r=0..7 are contiguous in n -> 16B store.
#pragma unroll
  for (int t = 0; t < 2; ++t) {
    const int d = (t == 0) ? dA : dB;
    union { unsigned short s[8]; uint4 q; } pk, pv;
#pragma unroll
    for (int r = 0; r < 8; ++r) {
      float kvv = (t == 0) ? kC0[r] : kC1[r];
      float vvv = (t == 0) ? vC0[r] : vC1[r];
      float ek = __expf(kvv);
      pk.s[r] = f2bfu(ek);
      pv.s[r] = f2bfu(ek * vvv);
    }
    size_t base = ((size_t)(b * 128 + d)) * NPAD + n0 + 8 * hi;
    *(uint4*)(expkT + base) = pk.q;
    *(uint4*)(ekvT + base) = pv.q;
  }

  // zero n in [2000,2048) so the K=2016 tail chunk multiplies against 0
  if (nt == 124) {
    for (int i = tid; i < 48 * 128; i += 128) {
      int n = 2000 + (i % 48), d = i / 48;
      size_t idx = ((size_t)(b * 128 + d)) * NPAD + n;
      expkT[idx] = 0;
      ekvT[idx] = 0;
    }
  }
}

// ---------------------------------------------------------------------------
// Kernel 2: q = q1@Wq1^T + q2@Wq2^T + last@Wq_last[:, :128]^T
//             + load*Wq_last[:,128] + left*Wq_last[:,129]; sigmoid -> sigq.
// grid (13, B), block 128.
// ---------------------------------------------------------------------------
__global__ void __launch_bounds__(128) prep_q_kernel(
    const float* __restrict__ q1, const float* __restrict__ q2,
    const float* __restrict__ lastn, const float* __restrict__ loadv,
    const float* __restrict__ leftv, const float* __restrict__ Wq1,
    const float* __restrict__ Wq2, const float* __restrict__ Wql,
    float* __restrict__ sigq) {
  const int pt = blockIdx.x, b = blockIdx.y;
  const int p0 = pt * 16;
  const int tid = threadIdx.x;
  const int lane = tid & 31, w = tid >> 5;
  const int col = lane & 15, hi = lane >> 4;

  int prow = p0 + col;
  if (prow > P_ - 1) prow = P_ - 1;  // clamp; stores are guarded
  const float* a1 = q1 + ((size_t)(b * P_ + prow)) * D_;
  const float* a2 = q2 + ((size_t)(b * P_ + prow)) * D_;
  const float* a3 = lastn + ((size_t)(b * P_ + prow)) * D_;

  const int dA = 2 * w * 16 + col, dB = dA + 16;
  v8f C0 = {}, C1 = {};
#pragma unroll
  for (int kc = 0; kc < 4; ++kc) {
    const int k0 = kc * 32;
    v16bf f1 = frag_f32(a1 + k0, hi);
    v16bf f2 = frag_f32(a2 + k0, hi);
    v16bf f3 = frag_f32(a3 + k0, hi);
    C0 = wmma_bf16(f1, frag_f32(Wq1 + (size_t)dA * D_ + k0, hi), C0);
    C0 = wmma_bf16(f2, frag_f32(Wq2 + (size_t)dA * D_ + k0, hi), C0);
    C0 = wmma_bf16(f3, frag_f32(Wql + (size_t)dA * 130 + k0, hi), C0);
    C1 = wmma_bf16(f1, frag_f32(Wq1 + (size_t)dB * D_ + k0, hi), C1);
    C1 = wmma_bf16(f2, frag_f32(Wq2 + (size_t)dB * D_ + k0, hi), C1);
    C1 = wmma_bf16(f3, frag_f32(Wql + (size_t)dB * 130 + k0, hi), C1);
  }

#pragma unroll
  for (int r = 0; r < 8; ++r) {
    const int p = p0 + r + 8 * hi;
    if (p < P_) {
      const float lw = loadv[b * P_ + p];
      const float lf = leftv[b * P_ + p];
      float qa = C0[r] + lw * Wql[(size_t)dA * 130 + 128] +
                 lf * Wql[(size_t)dA * 130 + 129];
      float qb = C1[r] + lw * Wql[(size_t)dB * 130 + 128] +
                 lf * Wql[(size_t)dB * 130 + 129];
      size_t base = ((size_t)(b * P_ + p)) * D_;
      sigq[base + dA] = 1.0f / (1.0f + __expf(-qa));
      sigq[base + dB] = 1.0f / (1.0f + __expf(-qb));
    }
  }
}

// ---------------------------------------------------------------------------
// Kernel 3: per (b, p-tile): bias/denom WMMA accumulation over N, AFT
// epilogue into LDS, score GEMM over 125 n-tiles, exp + deterministic
// row-sum softmax. grid (13, B), block 128 (4 waves).
// ---------------------------------------------------------------------------
__global__ void __launch_bounds__(128) aft_score_kernel(
    const float* __restrict__ cdist, const float* __restrict__ nmask,
    const float* __restrict__ sigq, const unsigned short* __restrict__ expkT,
    const unsigned short* __restrict__ ekvT,
    const unsigned short* __restrict__ nodesbf,
    const float* __restrict__ lscale, const float* __restrict__ aalpha,
    const float* __restrict__ palpha, float* __restrict__ out) {
  __shared__ float aft[16][D_];
  __shared__ float partial[4][32][8];
  __shared__ float rowsum[16];

  const int pt = blockIdx.x, b = blockIdx.y;
  const int p0 = pt * 16;
  const int tid = threadIdx.x;
  const int lane = tid & 31, w = tid >> 5;
  const int col = lane & 15, hi = lane >> 4;

  const float ls = lscale[0];
  const float laa = ls * aalpha[0];
  const float lpa = ls * palpha[0];

  int prow = p0 + col;
  if (prow > P_ - 1) prow = P_ - 1;
  const float* cdrow = cdist + ((size_t)(b * P_ + prow)) * N_;
  const float* nmrow = nmask + ((size_t)(b * P_ + prow)) * N_;

  const int dA = 2 * w * 16 + col, dB = dA + 16;
  const unsigned short* ekvA = ekvT + ((size_t)(b * 128 + dA)) * NPAD;
  const unsigned short* ekvB = ekvT + ((size_t)(b * 128 + dB)) * NPAD;
  const unsigned short* expA = expkT + ((size_t)(b * 128 + dA)) * NPAD;
  const unsigned short* expB = expkT + ((size_t)(b * 128 + dB)) * NPAD;

  // ---- phase 1: bias = e_alpha @ ekv, denom = e_alpha @ expk -------------
  v8f bC0 = {}, bC1 = {}, dC0 = {}, dC1 = {};
  for (int ch = 0; ch < 63; ++ch) {   // K = 63*32 = 2016 >= N (tail zeroed)
    const int n0 = ch * 32;
    // stream-ahead prefetch of the two HBM-resident input rows
    if (n0 + 256 + 32 <= N_) {
      __builtin_prefetch(cdrow + n0 + 256, 0, 1);   // global_prefetch_b8
      __builtin_prefetch(nmrow + n0 + 256, 0, 1);
    }
    v16bf a;
    {
      const int nb0 = n0 + 8 * hi;        // elements 0..7
      const int nb1 = n0 + 16 + 8 * hi;   // elements 8..15
      float dv[8], mv[8];
      if (nb0 + 7 < N_) {
        load8(cdrow + nb0, dv);
        load8(nmrow + nb0, mv);
#pragma unroll
        for (int i = 0; i < 8; ++i) a[i] = f2bf(__expf(laa * (-dv[i]) + mv[i]));
      } else {
#pragma unroll
        for (int i = 0; i < 8; ++i) a[i] = f2bf(0.0f);
      }
      if (nb1 + 7 < N_) {
        load8(cdrow + nb1, dv);
        load8(nmrow + nb1, mv);
#pragma unroll
        for (int i = 0; i < 8; ++i)
          a[8 + i] = f2bf(__expf(laa * (-dv[i]) + mv[i]));
      } else {
#pragma unroll
        for (int i = 0; i < 8; ++i) a[8 + i] = f2bf(0.0f);
      }
    }
    bC0 = wmma_bf16(a, frag_bf16(ekvA + n0, hi), bC0);
    dC0 = wmma_bf16(a, frag_bf16(expA + n0, hi), dC0);
    bC1 = wmma_bf16(a, frag_bf16(ekvB + n0, hi), bC1);
    dC1 = wmma_bf16(a, frag_bf16(expB + n0, hi), dC1);
  }

  // epilogue: aft_out = sigmoid_q * nan_to_num(bias)/(nan_to_num(denom)+eps)
#pragma unroll
  for (int r = 0; r < 8; ++r) {
    const int m = r + 8 * hi;
    int p = p0 + m;
    if (p > P_ - 1) p = P_ - 1;  // clamped rows hold finite junk, never stored
    const size_t sb = ((size_t)(b * P_ + p)) * D_;
#pragma unroll
    for (int t = 0; t < 2; ++t) {
      const int d = (t == 0) ? dA : dB;
      float bias = (t == 0) ? bC0[r] : bC1[r];
      float den = (t == 0) ? dC0[r] : dC1[r];
      if (bias != bias) bias = 0.0f;
      if (den != den) den = 0.0f;
      aft[m][d] = sigq[sb + d] * (bias / (den + 1e-20f));
    }
  }
  __syncthreads();

  // ---- phase 2: score = aft @ nodes^T, exp, deterministic softmax --------
  const float* arow = &aft[col][0];
  v16bf afr[4];
#pragma unroll
  for (int kc = 0; kc < 4; ++kc) afr[kc] = frag_f32(arow + kc * 32, hi);

  float myrs[8];
#pragma unroll
  for (int r = 0; r < 8; ++r) myrs[r] = 0.0f;

  for (int nt = w; nt < 125; nt += 4) {   // wave-uniform loop (WMMA-safe)
    const int ncol = nt * 16 + col;
    const unsigned short* brow = nodesbf + ((size_t)(b * N_ + ncol)) * D_;
    if (nt + 4 < 125) {                   // prefetch next tile's B rows
      __builtin_prefetch(nodesbf + ((size_t)(b * N_ + ncol + 64)) * D_, 0, 1);
    }
    v8f c = {};
#pragma unroll
    for (int kc = 0; kc < 4; ++kc)
      c = wmma_bf16(afr[kc], frag_bf16(brow + kc * 32, hi), c);
#pragma unroll
    for (int r = 0; r < 8; ++r) {
      const int p = p0 + r + 8 * hi;
      if (p < P_) {
        const size_t off = ((size_t)(b * P_ + p)) * N_ + ncol;
        float s = c[r] * INV_SQRT_D + lpa * (-cdist[off]);
        s = LOGIT_CLIP * tanhf(s) + nmask[off];
        const float e = __expf(s);   // bounded by e^10: max-free softmax safe
        out[off] = e;
        myrs[r] += e;
      }
    }
  }

  // deterministic cross-lane/cross-wave row-sum (fixed summation order)
#pragma unroll
  for (int r = 0; r < 8; ++r) partial[w][lane][r] = myrs[r];
  __threadfence();   // make out[] stores visible for the normalize pass
  __syncthreads();
  if (tid < 16) {
    const int rr = tid & 7, lo = (tid < 8) ? 0 : 16;
    float s = 0.0f;
    for (int ww = 0; ww < 4; ++ww)
      for (int l = 0; l < 16; ++l) s += partial[ww][lo + l][rr];
    rowsum[tid] = s;
  }
  __syncthreads();

  for (int i = tid; i < 16 * N_; i += 128) {
    const int m = i / N_, n = i - m * N_;
    const int p = p0 + m;
    if (p < P_) {
      const size_t off = ((size_t)(b * P_ + p)) * N_ + n;
      out[off] = out[off] / rowsum[m];
    }
  }
}

// ---------------------------------------------------------------------------
extern "C" void kernel_launch(void* const* d_in, const int* in_sizes, int n_in,
                              void* d_out, int out_size, void* d_ws,
                              size_t ws_size, hipStream_t stream) {
  (void)in_sizes; (void)n_in; (void)out_size; (void)ws_size;
  const float* nodes = (const float*)d_in[0];
  const float* lastn = (const float*)d_in[1];
  const float* q1    = (const float*)d_in[2];
  const float* q2    = (const float*)d_in[3];
  const float* loadv = (const float*)d_in[4];
  const float* leftv = (const float*)d_in[5];
  const float* cdist = (const float*)d_in[6];
  const float* nmask = (const float*)d_in[7];
  const float* lscal = (const float*)d_in[8];
  const float* Wq1   = (const float*)d_in[9];
  const float* Wq2   = (const float*)d_in[10];
  const float* Wql   = (const float*)d_in[11];
  const float* Wk    = (const float*)d_in[12];
  const float* Wv    = (const float*)d_in[13];
  const float* aalph = (const float*)d_in[14];
  const float* palph = (const float*)d_in[15];
  float* out = (float*)d_out;

  char* ws = (char*)d_ws;
  size_t off = 0;
  auto carve = [&](size_t bytes) -> void* {
    void* p = ws + off;
    off += (bytes + 255) & ~(size_t)255;
    return p;
  };
  unsigned short* expkT   = (unsigned short*)carve((size_t)B_ * 128 * NPAD * 2);
  unsigned short* ekvT    = (unsigned short*)carve((size_t)B_ * 128 * NPAD * 2);
  unsigned short* nodesbf = (unsigned short*)carve((size_t)B_ * N_ * D_ * 2);
  float*          sigq    = (float*)carve((size_t)B_ * P_ * D_ * 4);

  prep_kv_kernel<<<dim3(125, B_), 128, 0, stream>>>(nodes, Wk, Wv, expkT,
                                                    ekvT, nodesbf);
  prep_q_kernel<<<dim3(PTILES, B_), 128, 0, stream>>>(q1, q2, lastn, loadv,
                                                      leftv, Wq1, Wq2, Wql,
                                                      sigq);
  aft_score_kernel<<<dim3(PTILES, B_), 128, 0, stream>>>(
      cdist, nmask, sigq, expkT, ekvT, nodesbf, lscal, aalph, palph, out);
}
